// ResAttentionBlock_61521111548319
// MI455X (gfx1250) — matrix operand order repaired and checked by reading, compiled, not run on hardware
//
#include <hip/hip_runtime.h>
#include <hip/hip_bf16.h>

typedef __attribute__((ext_vector_type(16))) __bf16 v16bf;
typedef __attribute__((ext_vector_type(8)))  __bf16 v8bf;
typedef __attribute__((ext_vector_type(4)))  __bf16 v4bf;
typedef __attribute__((ext_vector_type(8)))  float  v8f;
typedef __attribute__((ext_vector_type(4)))  float  v4f;

#define HIDDEN      1024
#define LDS_STRIDE  1032   // 1024 + 8 bf16 pad -> 2064B row stride, conflict-free b128
#define MROWS       32     // rows per block (2 WMMA M-tiles per wave)
#define EPS_F       1e-6f

// ---------------- weight fp32 -> bf16 conversion (runs every launch) --------
__global__ __launch_bounds__(256) void cvt3_bf16(const float* __restrict__ a,
                                                 const float* __restrict__ b,
                                                 const float* __restrict__ c,
                                                 __bf16* __restrict__ out) {
    const float* src = (blockIdx.y == 0) ? a : (blockIdx.y == 1) ? b : c;
    __bf16* dst = out + (size_t)blockIdx.y * (HIDDEN * HIDDEN);
    int i = (blockIdx.x * 256 + threadIdx.x) * 4;
    v4f v = *(const v4f*)(src + i);
    v4bf o;
    o[0] = (__bf16)v[0]; o[1] = (__bf16)v[1];
    o[2] = (__bf16)v[2]; o[3] = (__bf16)v[3];
    *(v4bf*)(dst + i) = o;
}

// -------- fused GEMM helper: 32 rows (2 M-tiles) x 64 cols (4 N-tiles) -----
// acc[mt][t] ; B fragment reused across both M-tiles.
__device__ __forceinline__ void gemm24(const __bf16 (&sA)[MROWS][LDS_STRIDE],
                                       const __bf16* __restrict__ W,
                                       const float* __restrict__ bias,
                                       int cBase, int lr, int hi,
                                       v8f acc[2][4]) {
    const v8f z = {0.f,0.f,0.f,0.f,0.f,0.f,0.f,0.f};
#pragma unroll
    for (int mt = 0; mt < 2; ++mt)
#pragma unroll
        for (int t = 0; t < 4; ++t) acc[mt][t] = z;

    const int aoff = hi * 8;    // A: lane m holds K [k0+8hi,+8) and [k0+16+8hi,+8)
    const int boff = hi * 8;    // B: lane n holds K [k0+8hi,+8) and [k0+16+8hi,+8)
#pragma unroll 2
    for (int ks = 0; ks < HIDDEN; ks += 32) {
        v16bf af[2];
#pragma unroll
        for (int mt = 0; mt < 2; ++mt) {
            v8bf alo = *(const v8bf*)&sA[16*mt + lr][ks + aoff];
            v8bf ahi = *(const v8bf*)&sA[16*mt + lr][ks + 16 + aoff];
            af[mt] = __builtin_shufflevector(alo, ahi,
                        0,1,2,3,4,5,6,7,8,9,10,11,12,13,14,15);
        }
#pragma unroll
        for (int t = 0; t < 4; ++t) {
            const __bf16* bp = W + (size_t)(cBase + 16*t + lr) * HIDDEN + ks + boff;
            v8bf blo = *(const v8bf*)bp;
            v8bf bhi = *(const v8bf*)(bp + 16);
            v16bf bfr = __builtin_shufflevector(blo, bhi,
                        0,1,2,3,4,5,6,7,8,9,10,11,12,13,14,15);
#pragma unroll
            for (int mt = 0; mt < 2; ++mt)
                acc[mt][t] = __builtin_amdgcn_wmma_f32_16x16x32_bf16(
                    false, af[mt], false, bfr, (short)0, acc[mt][t], false, false);
        }
    }
#pragma unroll
    for (int t = 0; t < 4; ++t) {
        float bv = bias[cBase + 16*t + lr];
#pragma unroll
        for (int mt = 0; mt < 2; ++mt)
#pragma unroll
            for (int j = 0; j < 8; ++j) acc[mt][t][j] += bv;
    }
}

// ---------------- fused rmsnorm + 3x GEMM + cosine-attn + residual ----------
__global__ __launch_bounds__(512) void res_attn_fused(
        const float* __restrict__ x, const float* __restrict__ y,
        const int*   __restrict__ idx, const float* __restrict__ lnw,
        const __bf16* __restrict__ Wq, const float* __restrict__ qb,
        const __bf16* __restrict__ Wk, const float* __restrict__ kb,
        const __bf16* __restrict__ Wv, const float* __restrict__ vb,
        float* __restrict__ out) {
    __shared__ __align__(16) __bf16 sH[MROWS][LDS_STRIDE];  // rmsnorm(x[idx]) bf16
    __shared__ __align__(16) __bf16 sY[MROWS][LDS_STRIDE];  // y bf16

    const int tid  = threadIdx.x;
    const int wave = tid >> 5;          // 16 waves: wave == head index
    const int lane = tid & 31;
    const int lr   = lane & 15;
    const int hi   = lane >> 4;
    const int n0   = blockIdx.x * MROWS;

    // ---- Phase 1: stage activations (wave w handles rows 2w, 2w+1) ----
#pragma unroll
    for (int rr = 0; rr < 2; ++rr) {
        const int r = wave * 2 + rr;
        const size_t gq = (size_t)idx[n0 + r] * HIDDEN;
        const size_t gy = (size_t)(n0 + r) * HIDDEN;
        float xv[32];
        float ssq = 0.f;
#pragma unroll
        for (int j = 0; j < 32; ++j) {
            float v = x[gq + lane + 32*j];
            xv[j] = v; ssq += v * v;
        }
#pragma unroll
        for (int m = 16; m >= 1; m >>= 1) ssq += __shfl_xor(ssq, m, 32);
        const float scale = rsqrtf(ssq * (1.0f / HIDDEN) + EPS_F);
#pragma unroll
        for (int j = 0; j < 32; ++j) {
            const int c = lane + 32*j;
            sH[r][c] = (__bf16)(xv[j] * scale * lnw[c]);
            sY[r][c] = (__bf16)y[gy + c];
        }
    }
    __syncthreads();

    // ---- Phase 2: Q and K GEMMs (wave owns head `wave`: cols [64w, 64w+64)) ----
    const int cBase = wave * 64;
    v8f qacc[2][4], kacc[2][4];
    gemm24(sH, Wq, qb, cBase, lr, hi, qacc);
    gemm24(sY, Wk, kb, cBase, lr, hi, kacc);

    // ---- Phase 3: per-row cosine similarity for this wave's head ----
    // C layout: VGPR j, lanes 0-15 -> row j ; lanes 16-31 -> row j+8.
    // The head's 64 cols = 4 tiles x 16 lanes of each half.
    float att[2][8];
#pragma unroll
    for (int mt = 0; mt < 2; ++mt) {
#pragma unroll
        for (int j = 0; j < 8; ++j) {
            float d = 0.f, qq = 0.f, kk = 0.f;
#pragma unroll
            for (int t = 0; t < 4; ++t) {
                float qv = qacc[mt][t][j], kv = kacc[mt][t][j];
                d += qv * kv; qq += qv * qv; kk += kv * kv;
            }
#pragma unroll
            for (int m = 8; m >= 1; m >>= 1) {   // reduce within 16-lane half
                d  += __shfl_xor(d,  m, 32);
                qq += __shfl_xor(qq, m, 32);
                kk += __shfl_xor(kk, m, 32);
            }
            const float qn = fmaxf(sqrtf(qq), EPS_F);
            const float kn = fmaxf(sqrtf(kk), EPS_F);
            att[mt][j] = d / (qn * kn);
        }
    }

    // ---- Phase 4: V GEMM (q/k accumulators now dead -> regs reused) ----
    v8f vacc[2][4];
    gemm24(sY, Wv, vb, cBase, lr, hi, vacc);

    // ---- Phase 5: out = x + v * att ----
#pragma unroll
    for (int mt = 0; mt < 2; ++mt) {
#pragma unroll
        for (int t = 0; t < 4; ++t) {
            const int c = cBase + 16*t + lr;
#pragma unroll
            for (int j = 0; j < 8; ++j) {
                const int m = 16*mt + j + 8*hi;
                const size_t g = (size_t)(n0 + m) * HIDDEN + c;
                out[g] = x[g] + vacc[mt][t][j] * att[mt][j];
            }
        }
    }
}

extern "C" void kernel_launch(void* const* d_in, const int* in_sizes, int n_in,
                              void* d_out, int out_size, void* d_ws, size_t ws_size,
                              hipStream_t stream) {
    const float* x   = (const float*)d_in[0];
    const float* y   = (const float*)d_in[1];
    const int*   idx = (const int*)  d_in[2];
    const float* lnw = (const float*)d_in[3];
    const float* qw  = (const float*)d_in[4];
    const float* qb  = (const float*)d_in[5];
    const float* kw  = (const float*)d_in[6];
    const float* kb  = (const float*)d_in[7];
    const float* vw  = (const float*)d_in[8];
    const float* vb  = (const float*)d_in[9];
    float* out = (float*)d_out;

    __bf16* wbf = (__bf16*)d_ws;                 // 3 x 2MB bf16 weights
    __bf16* Wq = wbf;
    __bf16* Wk = wbf + (size_t)HIDDEN * HIDDEN;
    __bf16* Wv = wbf + (size_t)2 * HIDDEN * HIDDEN;

    const int N = in_sizes[0] / HIDDEN;

    dim3 cgrid((HIDDEN * HIDDEN) / (256 * 4), 3);
    cvt3_bf16<<<cgrid, 256, 0, stream>>>(qw, kw, vw, wbf);

    res_attn_fused<<<N / MROWS, 512, 0, stream>>>(
        x, y, idx, lnw, Wq, qb, Wk, kb, Wv, vb, out);
}
